// NashRouter_74088185856328
// MI455X (gfx1250) — matrix operand order repaired and checked by reading, not compiled, not run on hardware
//
#include <hip/hip_runtime.h>
#include <math.h>

// ---------------------------------------------------------------------------
// NashRouter for MI455X (gfx1250, wave32).
//
// Roofline: x stream = 128 MB -> ~5.5 us @ 23.3 TB/s. 4.3 GFLOP at AI ~32
// FLOP/B needs ~746 TF, so f32 WMMA (16x16x4, 8x less FLOP/issue) would be
// compute-bound; we convert in-flight to f16 (v_cvt_pk_f16_f32) and use
// v_wmma_f32_16x16x32_f16 with f32 accumulation -> HBM-bound.
//
// B-operand strategy: all 8 waves of a block share the same W, so Wh (f16,
// pre-converted once into d_ws) is staged into LDS in 32 KB K-chunks and B
// fragments are built with ds_load_b128. This cuts ~256 MB of per-wave L2
// B-traffic (which would demand ~47 TB/s) down to ~32 MB of staging traffic,
// moving the steady-state B feed onto WGP-private LDS banks.
// ---------------------------------------------------------------------------

typedef _Float16 v16h __attribute__((ext_vector_type(16)));
typedef _Float16 v8h  __attribute__((ext_vector_type(8)));
typedef float    v8f  __attribute__((ext_vector_type(8)));
typedef float    v4f  __attribute__((ext_vector_type(4)));

#define D_DIM 1024
#define E_DIM 64
#define TOK_PER_BLOCK 128   // 8 waves * 16 tokens (WMMA M)
#define NWAVES 8
#define KCHUNK 256          // K-elements staged per round (32 KB of f16)
#define WROW   (KCHUNK + 8) // +16B pad: expert rows land on distinct bank groups

// Prep: W f32 -> f16 scratch (one-time, L2-resident); zero usage output
// (d_out is poisoned with 0xAA before timing).
__global__ void __launch_bounds__(256)
nash_prep_kernel(const float* __restrict__ W, _Float16* __restrict__ Wh,
                 float* __restrict__ outU) {
  int i = blockIdx.x * 256 + threadIdx.x;
  if (i < E_DIM * D_DIM) Wh[i] = (_Float16)W[i];
  if (i < E_DIM) outU[i] = 0.0f;
}

__global__ void __launch_bounds__(256)
nash_router_kernel(const float* __restrict__ x,
                   const _Float16* __restrict__ Wh,
                   const float* __restrict__ temperature,
                   const float* __restrict__ regret,
                   const int* __restrict__ step_count,
                   float* __restrict__ outW,
                   int* __restrict__ outI,
                   float* __restrict__ outU,
                   int nTokens) {
  // sW: staged W chunk [64 experts][KCHUNK halves], padded row stride.
  __shared__ _Float16 sW[E_DIM * WROW];      // 33,792 B
  // util row stride 65: lane halves write rows m and m+8 -> bank shift of 8,
  // conflict-free ds_store of accumulators.
  __shared__ float util[NWAVES][16][65];     // 33,280 B
  __shared__ float s_lp[E_DIM];
  __shared__ float s_usage[E_DIM];

  const int tid  = threadIdx.x;
  const int wave = tid >> 5;
  const int lane = tid & 31;
  const int m    = lane & 15;   // M row (A) / N col (B) within 16-wide tile
  const int kh   = lane >> 4;   // lane-half selects K sub-range (ISA layout)

  // Per-block scalar prep: regret-matching log prior + usage counters.
  if (tid < E_DIM) {
    s_usage[tid] = 0.0f;
    float lp = 0.0f;
    if (step_count[0] >= 100) {
      float tot = 0.0f;
      for (int e = 0; e < E_DIM; ++e) tot += fmaxf(regret[e], 0.0f);
      float pos   = fmaxf(regret[tid], 0.0f);
      float prior = (tot > 0.0f) ? pos / fmaxf(tot, 1e-30f)
                                 : (1.0f / (float)E_DIM);
      lp = logf(prior + 1e-8f);
    }
    s_lp[tid] = lp;
  }

  const int tokBase = blockIdx.x * TOK_PER_BLOCK + wave * 16;
  const float* xrow = x + (size_t)(tokBase + m) * D_DIM;

  // Staging role: thread j moves one 128 B sub-run of one expert row.
  const int se = tid >> 2;        // expert row 0..63
  const int sq = tid & 3;         // which 64-half sub-run

  v8f z = {0, 0, 0, 0, 0, 0, 0, 0};
  v8f acc[4] = {z, z, z, z};

  for (int c = 0; c < D_DIM / KCHUNK; ++c) {
    const int k0 = c * KCHUNK;

    __syncthreads();  // previous chunk fully consumed (also orders s_lp init)
    {
      const v8h* src = (const v8h*)(Wh + (size_t)se * D_DIM + k0 + sq * 64);
      _Float16* dst = &sW[se * WROW + sq * 64];
#pragma unroll
      for (int i = 0; i < 8; ++i)
        *(v8h*)(dst + i * 8) = src[i];   // global_load_b128 -> ds_store_b128
    }
    __syncthreads();  // staged chunk visible to all waves

    for (int kk = k0; kk < k0 + KCHUNK; kk += 32) {
      // A fragment (16x32 f16): lane holds K = kk+8*kh+{0..7} and +16..+23:
      // two contiguous 8-float runs, 4x NT global_load_b128, packed cvt.
      const v4f* xp = (const v4f*)(xrow + kk + 8 * kh);
      v4f x0 = __builtin_nontemporal_load(xp + 0);
      v4f x1 = __builtin_nontemporal_load(xp + 1);
      v4f x2 = __builtin_nontemporal_load(xp + 4);
      v4f x3 = __builtin_nontemporal_load(xp + 5);
      v16h a;
#pragma unroll
      for (int i = 0; i < 4; ++i) {
        a[i]      = (_Float16)x0[i];
        a[i + 4]  = (_Float16)x1[i];
        a[i + 8]  = (_Float16)x2[i];
        a[i + 12] = (_Float16)x3[i];
      }

      // 4 expert tiles: B fragment (32x16 f16): lane = column e, K halves
      // kk+8*kh+{0..7} and +16..+23 -> two ds_load_b128 from LDS.
#pragma unroll
      for (int t = 0; t < 4; ++t) {
        const _Float16* bp = &sW[(t * 16 + m) * WROW + (kk - k0) + 8 * kh];
        v8h b0 = *(const v8h*)bp;
        v8h b1 = *(const v8h*)(bp + 16);
        v16h b;
#pragma unroll
        for (int i = 0; i < 8; ++i) { b[i] = b0[i]; b[i + 8] = b1[i]; }
        acc[t] = __builtin_amdgcn_wmma_f32_16x16x32_f16(
            /*neg_a=*/false, a, /*neg_b=*/false, b,
            /*c_mod=*/(short)0, acc[t], /*reuse_a=*/false, /*reuse_b=*/false);
      }
    }
  }

  // C layout: acc[t][r] is (M = r + 8*kh, N = t*16 + m). Spill tile to LDS.
#pragma unroll
  for (int t = 0; t < 4; ++t)
#pragma unroll
    for (int r = 0; r < 8; ++r)
      util[wave][r + 8 * kh][t * 16 + m] = acc[t][r];

  __syncthreads();

  // ------------------- Epilogue: one thread per token ------------------------
  if (tid < TOK_PER_BLOCK) {
    const int w2 = tid >> 4, m2 = tid & 15;
    const int tok = blockIdx.x * TOK_PER_BLOCK + tid;
    const float invT = 1.0f / fmaxf(fabsf(temperature[0]), 0.1f);

    // Pass 1: top-2 (stable: earlier index wins ties, matches jax top_k).
    float m1 = -INFINITY, mB = -INFINITY;
    int i1 = 0, i2 = 0;
    for (int e = 0; e < E_DIM; ++e) {
      float v = util[w2][m2][e] * invT + s_lp[e];
      if (v > m1)      { mB = m1; i2 = i1; m1 = v; i1 = e; }
      else if (v > mB) { mB = v; i2 = e; }
    }
    // Pass 2: softmax partition function.
    float Z = 0.0f;
    for (int e = 0; e < E_DIM; ++e) {
      float v = util[w2][m2][e] * invT + s_lp[e];
      Z += __expf(v - m1);
    }
    float p1 = 1.0f / Z;
    float p2 = __expf(mB - m1) / Z;
    float s  = p1 + p2 + 1e-8f;
    outW[tok * 2 + 0] = p1 / s;
    outW[tok * 2 + 1] = p2 / s;
    outI[tok * 2 + 0] = i1;
    outI[tok * 2 + 1] = i2;
    atomicAdd(&s_usage[i1], 1.0f);   // ds_add_f32
    atomicAdd(&s_usage[i2], 1.0f);
  }
  __syncthreads();

  if (tid < E_DIM) {
    float c = s_usage[tid];
    if (c != 0.0f)
      atomicAdd(&outU[tid], c * (1.0f / (2.0f * (float)nTokens)));
  }
}

extern "C" void kernel_launch(void* const* d_in, const int* in_sizes, int n_in,
                              void* d_out, int out_size, void* d_ws, size_t ws_size,
                              hipStream_t stream) {
  const float* x           = (const float*)d_in[0];  // [B,L,D] f32
  const float* W           = (const float*)d_in[1];  // [E,D]   f32
  const float* temperature = (const float*)d_in[2];  // [1]     f32
  const float* regret      = (const float*)d_in[3];  // [E]     f32
  const int*   step        = (const int*)d_in[4];    // scalar int

  const int nTokens = in_sizes[0] / D_DIM;           // 32768

  // Output: weights f32 [2*nTokens] | indices i32 [2*nTokens] | usage f32 [E]
  float* outW = (float*)d_out;
  int*   outI = (int*)((float*)d_out + 2 * nTokens);
  float* outU = (float*)d_out + 4 * nTokens;

  _Float16* Wh = (_Float16*)d_ws;                    // needs 128 KB scratch

  nash_prep_kernel<<<(E_DIM * D_DIM + 255) / 256, 256, 0, stream>>>(W, Wh, outU);
  nash_router_kernel<<<nTokens / TOK_PER_BLOCK, 256, 0, stream>>>(
      x, Wh, temperature, regret, step, outW, outI, outU, nTokens);
}
